// BlqSSM_40381282517432
// MI455X (gfx1250) — matrix operand order repaired and checked by reading, not compiled, not run on hardware
//
#include <hip/hip_runtime.h>
#include <hip/hip_bf16.h>

#define DIMC 256
#define DSTATE 64
#define NHEADS 8
#define HEAD_DIM 32
#define CONV_CH 384
#define D_IN_PROJ 648
#define NPROJ_PAD 656      // padded to 41 tiles of 16
#define LSEQ 4096
#define BATCH 8
#define HH 64
#define WW 64
#define LN_EPS 1e-5f

// gfx1250 async global->LDS staging (inline asm; verified assembling in round 2)
#ifndef USE_ASYNC_LDS
#define USE_ASYNC_LDS 1
#endif

typedef __attribute__((ext_vector_type(16))) _Float16 v16h;
typedef __attribute__((ext_vector_type(8)))  _Float16 v8h;
typedef __attribute__((ext_vector_type(4)))  _Float16 v4h;
typedef __attribute__((ext_vector_type(8)))  float    v8f;
typedef __attribute__((ext_vector_type(4)))  float    v4f;

// ---- WMMA fragment builders (layouts per CDNA5 ISA §7.12.2) ----
// A (16x32 f16): lanes hold row M=lane&15; element e -> K=kb+hf*16+e
// (contiguous, matching B).
static __device__ __forceinline__ v16h wmma_a_load(const _Float16* rowptr, int kb, int hf) {
  const _Float16* p = rowptr + kb + hf * 16;
  v8h lo = *(const v8h*)(p);
  v8h hi = *(const v8h*)(p + 8);
  v16h r;
#pragma unroll
  for (int i = 0; i < 8; ++i) { r[i] = lo[i]; r[i + 8] = hi[i]; }
  return r;
}
// B (32x16 f16): lane holds column N; element e -> K = kb + hf*16 + e (contiguous).
// For C = X * W^T with row-major W, column N of B == row N of W: one v16h load.
static __device__ __forceinline__ v16h wmma_b_load_row(const _Float16* rowptr, int kb, int hf) {
  return *(const v16h*)(rowptr + kb + hf * 16);
}

static __device__ __forceinline__ float softplus_f(float x) {
  return (x > 20.f) ? x : log1pf(expf(x));
}
static __device__ __forceinline__ float silu_f(float x) {
  return x / (1.f + expf(-x));
}

// ---- 1. transpose x (B,C,L) -> Xh (B,L,C) f16, LDS tiled ----
__global__ void k_transpose_x(const float* __restrict__ x, _Float16* __restrict__ Xh) {
  __shared__ float tile[32][33];
  int b = blockIdx.z;
  int c0 = blockIdx.y * 32;
  int l0 = blockIdx.x * 32;
  int t = threadIdx.x;
  int tr = t >> 5, tc = t & 31;
#pragma unroll
  for (int i = 0; i < 4; ++i) {
    int c = c0 + tr + i * 8;
    tile[tr + i * 8][tc] = x[((size_t)b * DIMC + c) * LSEQ + l0 + tc];
  }
  __syncthreads();
#pragma unroll
  for (int i = 0; i < 4; ++i) {
    int l = l0 + tr + i * 8;
    Xh[((size_t)b * LSEQ + l) * DIMC + c0 + tc] = (_Float16)tile[tc][tr + i * 8];
  }
}

// ---- 2. convert weights to f16 (W_in padded with zero rows to 656) ----
__global__ void k_cvt_weights(const float* __restrict__ W_in, const float* __restrict__ W_out,
                              _Float16* __restrict__ Winh, _Float16* __restrict__ Woh) {
  int idx = blockIdx.x * blockDim.x + threadIdx.x;
  if (idx < NPROJ_PAD * DIMC) {
    int r = idx / DIMC;
    Winh[idx] = (r < D_IN_PROJ) ? (_Float16)W_in[idx] : (_Float16)0.f;
  } else {
    int j = idx - NPROJ_PAD * DIMC;
    if (j < DIMC * DIMC) Woh[j] = (_Float16)W_out[j];
  }
}

// ---- 3. in-proj GEMM (L x 256) @ (256 x 648), tile-uniform output routing ----
__global__ void k_inproj(const _Float16* __restrict__ Xh, const _Float16* __restrict__ Winh,
                         const float* __restrict__ A_log, const float* __restrict__ dt_bias,
                         _Float16* __restrict__ zh, _Float16* __restrict__ xBCr,
                         float* __restrict__ dtA) {
  int b = blockIdx.y;
  int l0 = blockIdx.x * 16;
  int wave = __builtin_amdgcn_readfirstlane(threadIdx.x >> 5);  // force SGPR: scalar routing
  int lane = threadIdx.x & 31;
  int hf = lane >> 4, ln = lane & 15;

  const _Float16* arow = Xh + ((size_t)b * LSEQ + l0 + ln) * DIMC;
  v16h afr[8];
#pragma unroll
  for (int k = 0; k < 8; ++k) afr[k] = wmma_a_load(arow, k * 32, hf);

  // one-shot prefetch of this wave's second weight tile (global_prefetch_b8)
  __builtin_prefetch(Winh + (size_t)((wave + 8) * 16 + ln) * DIMC, 0, 3);

  for (int nt = wave; nt < (NPROJ_PAD / 16); nt += 8) {
    const _Float16* brow = Winh + (size_t)(nt * 16 + ln) * DIMC;
    v8f acc = {};
    v16h bcur = wmma_b_load_row(brow, 0, hf);
#pragma unroll
    for (int k = 0; k < 8; ++k) {
      v16h bnxt = bcur;
      if (k < 7) bnxt = wmma_b_load_row(brow, (k + 1) * 32, hf);  // overlap with WMMA
      acc = __builtin_amdgcn_wmma_f32_16x16x32_f16(false, afr[k], false, bcur,
                                                   (short)0, acc, false, false);
      bcur = bnxt;
    }
    // routing is uniform per n-tile (scalar branch): [0,16)->z, [16,40)->xBC, 40->dt
    if (nt < 16) {
      _Float16* dst = zh + ((size_t)b * LSEQ + l0) * DIMC + nt * 16 + ln;
#pragma unroll
      for (int r = 0; r < 8; ++r)
        dst[(size_t)(r + hf * 8) * DIMC] = (_Float16)acc[r];
    } else if (nt < 40) {
      _Float16* dst = xBCr + ((size_t)b * LSEQ + l0) * CONV_CH + (nt * 16 - DIMC) + ln;
#pragma unroll
      for (int r = 0; r < 8; ++r)
        dst[(size_t)(r + hf * 8) * CONV_CH] = (_Float16)acc[r];
    } else if (ln < NHEADS) {
      int h = ln;
      float ea = expf(A_log[h]);
      float db = dt_bias[h];
      float* dst = dtA + ((size_t)b * LSEQ + l0) * NHEADS + h;
#pragma unroll
      for (int r = 0; r < 8; ++r)
        dst[(size_t)(r + hf * 8) * NHEADS] = softplus_f(acc[r] + db) * ea;
    }
  }
}

// ---- 4. depthwise 3x3 conv + bias + SiLU ----
__global__ void k_conv(const _Float16* __restrict__ xBCr, const float* __restrict__ conv_w,
                       const float* __restrict__ conv_b, _Float16* __restrict__ xBCc) {
  int b = blockIdx.y;
  int l = blockIdx.x;
  int c = threadIdx.x;
  int hh = l >> 6, ww = l & 63;
  float acc = 0.f;
#pragma unroll
  for (int dy = 0; dy < 3; ++dy) {
    int y = hh + dy - 1;
    if ((unsigned)y >= HH) continue;
#pragma unroll
    for (int dx = 0; dx < 3; ++dx) {
      int xw = ww + dx - 1;
      if ((unsigned)xw >= WW) continue;
      float w = conv_w[c * 9 + dy * 3 + dx];
      acc += w * (float)xBCr[((size_t)b * LSEQ + y * WW + xw) * CONV_CH + c];
    }
  }
  acc += conv_b[c];
  xBCc[((size_t)b * LSEQ + l) * CONV_CH + c] = (_Float16)silu_f(acc);
}

__global__ void k_zero(float* __restrict__ p, int n) {
  int i = blockIdx.x * blockDim.x + threadIdx.x;
  if (i < n) p[i] = 0.f;
}

// ---- 5. KV[s,p] = sum_l Bm[l,s] * (V*dA)[l,p]  (split-K + f32 atomics) ----
__global__ void k_kv(const _Float16* __restrict__ xBCc, const float* __restrict__ dtA,
                     float* __restrict__ KVf) {
  __shared__ _Float16 ldsBt[DSTATE * 40];    // [s][l'] stride 40 halfs (16B aligned rows)
  __shared__ _Float16 ldsVt[HEAD_DIM * 40];  // [p][l'] stride 40 halfs
  int b = blockIdx.z, h = blockIdx.y;
  int lchunk = blockIdx.x * 512;
  int t = threadIdx.x;
  int wave = __builtin_amdgcn_readfirstlane(t >> 5);
  int lane = t & 31, hf = lane >> 4, ln = lane & 15;
  int stile = wave >> 1, ptile = wave & 1;

  v8f acc = {};
  for (int it = 0; it < 16; ++it) {
    int lb = lchunk + it * 32;
    __syncthreads();
    {
      int lp = t >> 3;
      size_t rowoff = (size_t)b * LSEQ + lb + lp;
      int s8 = (t & 7) * 8;
      v8h v = *(const v8h*)(xBCc + rowoff * CONV_CH + DIMC + s8);
#pragma unroll
      for (int i = 0; i < 8; ++i) ldsBt[(s8 + i) * 40 + lp] = v[i];
      int p4 = (t & 7) * 4;
      float da = dtA[rowoff * NHEADS + h];
      v4h u = *(const v4h*)(xBCc + rowoff * CONV_CH + h * HEAD_DIM + p4);
#pragma unroll
      for (int i = 0; i < 4; ++i) ldsVt[(p4 + i) * 40 + lp] = (_Float16)((float)u[i] * da);
    }
    __syncthreads();
    int srow = stile * 16 + ln;
    v8h lo = *(const v8h*)&ldsBt[srow * 40 + hf * 16];
    v8h hi = *(const v8h*)&ldsBt[srow * 40 + hf * 16 + 8];
    v16h a;
#pragma unroll
    for (int i = 0; i < 8; ++i) { a[i] = lo[i]; a[i + 8] = hi[i]; }
    int prow = ptile * 16 + ln;
    v16h bfr = *(const v16h*)&ldsVt[prow * 40 + hf * 16];
    acc = __builtin_amdgcn_wmma_f32_16x16x32_f16(false, a, false, bfr,
                                                 (short)0, acc, false, false);
  }
  int p = ptile * 16 + ln;
#pragma unroll
  for (int r = 0; r < 8; ++r) {
    int s = stile * 16 + r + hf * 8;
    atomicAdd(&KVf[(((size_t)b * NHEADS + h) * DSTATE + s) * HEAD_DIM + p], acc[r]);
  }
}

// ---- 5b. convert + transpose KV to per-head [p][s] f16 so k_y B-frags are
//          contiguous 32B vector loads ----
__global__ void k_kvcvt_t(const float* __restrict__ KVf, _Float16* __restrict__ KVt, int n) {
  int i = blockIdx.x * blockDim.x + threadIdx.x;
  if (i < n) {
    int p = i & (HEAD_DIM - 1);
    int s = (i >> 5) & (DSTATE - 1);
    int bh = i >> 11;  // DSTATE*HEAD_DIM == 2048
    KVt[((size_t)bh * HEAD_DIM + p) * DSTATE + s] = (_Float16)KVf[i];
  }
}

// ---- 6. y = Cm @ KV + V*D, fused LayerNorm + z-gate ----
#define YSTR 264
__global__ void k_y(const _Float16* __restrict__ xBCc, const _Float16* __restrict__ KVt,
                    const float* __restrict__ Dv, const float* __restrict__ ln_w,
                    const float* __restrict__ ln_b, const _Float16* __restrict__ zh,
                    _Float16* __restrict__ ygh) {
  __shared__ float ytile[16 * YSTR];
  __shared__ float psum[16][16], psq[16][16];
  __shared__ float rowmu[16], rowrs[16];
  int b = blockIdx.y;
  int l0 = blockIdx.x * 16;
  int t = threadIdx.x;
  int wave = __builtin_amdgcn_readfirstlane(t >> 5);
  int lane = t & 31, hf = lane >> 4, ln = lane & 15;
  int h = wave;  // one head per wave (scalar)
  float Dh = Dv[h];
  const _Float16* crow = xBCc + ((size_t)b * LSEQ + l0 + ln) * CONV_CH + DIMC + DSTATE;

#if USE_ASYNC_LDS
  // Stage this head's 32x64 f16 KV^T tile into LDS via async global->LDS copy
  // (GLOBAL_LOAD_ASYNC_TO_LDS_B128, ASYNCcnt-tracked; ISA 15.18.3 op 98).
  __shared__ __align__(16) _Float16 kvlds[NHEADS * DSTATE * HEAD_DIM];
  {
    const _Float16* gsrc = KVt + ((size_t)b * NHEADS + h) * DSTATE * HEAD_DIM;
    unsigned int lbase = (unsigned int)(size_t)&kvlds[h * DSTATE * HEAD_DIM];
#pragma unroll
    for (int i = 0; i < 8; ++i) {
      unsigned int boff = (unsigned int)(i * 32 + lane) * 16u;
      unsigned long long ga = (unsigned long long)(size_t)gsrc + boff;
      unsigned int la = lbase + boff;
      asm volatile("global_load_async_to_lds_b128 %0, %1, off"
                   :: "v"(la), "v"(ga) : "memory");
    }
    asm volatile("s_wait_asynccnt 0" ::: "memory");
  }
  const _Float16* kvb = &kvlds[h * DSTATE * HEAD_DIM];  // [p][s], per-wave tile
#else
  const _Float16* kvb = KVt + ((size_t)b * NHEADS + h) * DSTATE * HEAD_DIM;
#endif

#pragma unroll
  for (int pt = 0; pt < 2; ++pt) {
    int p = pt * 16 + ln;
    v8f acc = {};
#pragma unroll
    for (int kk = 0; kk < 2; ++kk) {
      v16h a = wmma_a_load(crow, kk * 32, hf);
      // B[k=s][n=p]: lane p reads KV^T row p, 16 contiguous s values
      v16h bfr = *(const v16h*)(kvb + p * DSTATE + kk * 32 + hf * 16);
      acc = __builtin_amdgcn_wmma_f32_16x16x32_f16(false, a, false, bfr,
                                                   (short)0, acc, false, false);
    }
#pragma unroll
    for (int r = 0; r < 8; ++r) {
      int m = r + hf * 8;
      int l = l0 + m;
      float vres = (float)xBCc[((size_t)b * LSEQ + l) * CONV_CH + h * HEAD_DIM + p];
      ytile[m * YSTR + h * HEAD_DIM + p] = acc[r] + vres * Dh;
    }
  }
  __syncthreads();
  int row = t >> 4, seg = t & 15;
  float s = 0.f, sq = 0.f;
#pragma unroll
  for (int i = 0; i < 16; ++i) {
    float v = ytile[row * YSTR + seg * 16 + i];
    s += v; sq += v * v;
  }
  psum[row][seg] = s; psq[row][seg] = sq;
  __syncthreads();
  if (seg == 0) {
    float ts = 0.f, tq = 0.f;
#pragma unroll
    for (int i = 0; i < 16; ++i) { ts += psum[row][i]; tq += psq[row][i]; }
    float mu = ts * (1.f / 256.f);
    float var = tq * (1.f / 256.f) - mu * mu;
    rowmu[row] = mu;
    rowrs[row] = rsqrtf(var + LN_EPS);
  }
  __syncthreads();
  float mu = rowmu[row], rs = rowrs[row];
  int l = l0 + row;
#pragma unroll
  for (int i = 0; i < 16; ++i) {
    int c = seg * 16 + i;
    float v = (ytile[row * YSTR + c] - mu) * rs * ln_w[c] + ln_b[c];
    v *= (float)zh[((size_t)b * LSEQ + l) * DIMC + c];
    ygh[((size_t)b * LSEQ + l) * DIMC + c] = (_Float16)v;
  }
}

// ---- 7. out = yg @ W_out^T, stored as (B, C, L) via per-wave LDS transpose ----
__global__ void k_out(const _Float16* __restrict__ ygh, const _Float16* __restrict__ Woh,
                      float* __restrict__ out) {
  __shared__ float otile[8][16][17];
  int b = blockIdx.y;
  int l0 = blockIdx.x * 16;
  int t = threadIdx.x;
  int wave = __builtin_amdgcn_readfirstlane(t >> 5);
  int lane = t & 31, hf = lane >> 4, ln = lane & 15;
  const _Float16* arow = ygh + ((size_t)b * LSEQ + l0 + ln) * DIMC;
  v16h afr[8];
#pragma unroll
  for (int k = 0; k < 8; ++k) afr[k] = wmma_a_load(arow, k * 32, hf);
  __builtin_prefetch(Woh + (size_t)((wave * 2 + 1) * 16 + ln) * DIMC, 0, 3);
#pragma unroll
  for (int tix = 0; tix < 2; ++tix) {
    int nt = wave * 2 + tix;
    const _Float16* brow = Woh + (size_t)(nt * 16 + ln) * DIMC;
    v8f acc = {};
    v16h bcur = wmma_b_load_row(brow, 0, hf);
#pragma unroll
    for (int k = 0; k < 8; ++k) {
      v16h bnxt = bcur;
      if (k < 7) bnxt = wmma_b_load_row(brow, (k + 1) * 32, hf);
      acc = __builtin_amdgcn_wmma_f32_16x16x32_f16(false, afr[k], false, bcur,
                                                   (short)0, acc, false, false);
      bcur = bnxt;
    }
    __syncthreads();
#pragma unroll
    for (int r = 0; r < 8; ++r) otile[wave][r + hf * 8][ln] = acc[r];
    __syncthreads();
    int nrow = lane >> 1, colh = lane & 1;
    float vals[8];
#pragma unroll
    for (int i = 0; i < 8; ++i) vals[i] = otile[wave][colh * 8 + i][nrow];
    size_t base = ((size_t)b * DIMC + nt * 16 + nrow) * LSEQ + l0 + colh * 8;
    *(v4f*)(out + base)     = *(const v4f*)&vals[0];
    *(v4f*)(out + base + 4) = *(const v4f*)&vals[4];
  }
}

extern "C" void kernel_launch(void* const* d_in, const int* in_sizes, int n_in,
                              void* d_out, int out_size, void* d_ws, size_t ws_size,
                              hipStream_t stream) {
  const float* x       = (const float*)d_in[0];
  const float* W_in    = (const float*)d_in[1];
  const float* conv_w  = (const float*)d_in[2];
  const float* conv_b  = (const float*)d_in[3];
  const float* A_log   = (const float*)d_in[4];
  const float* dt_bias = (const float*)d_in[5];
  const float* Dv      = (const float*)d_in[6];
  const float* ln_w    = (const float*)d_in[7];
  const float* ln_b    = (const float*)d_in[8];
  const float* W_out   = (const float*)d_in[9];
  float* out = (float*)d_out;

  char* ws = (char*)d_ws;
  size_t off = 0;
  auto carve = [&](size_t bytes) -> char* {
    char* p = ws + off;
    off += (bytes + 255) & ~(size_t)255;
    return p;
  };
  _Float16* Xh   = (_Float16*)carve((size_t)BATCH * LSEQ * DIMC * 2);
  _Float16* Winh = (_Float16*)carve((size_t)NPROJ_PAD * DIMC * 2);
  _Float16* Woh  = (_Float16*)carve((size_t)DIMC * DIMC * 2);
  _Float16* zh   = (_Float16*)carve((size_t)BATCH * LSEQ * DIMC * 2);
  _Float16* xBCr = (_Float16*)carve((size_t)BATCH * LSEQ * CONV_CH * 2);
  float*    dtA  = (float*)   carve((size_t)BATCH * LSEQ * NHEADS * 4);
  _Float16* xBCc = (_Float16*)carve((size_t)BATCH * LSEQ * CONV_CH * 2);
  float*    KVf  = (float*)   carve((size_t)BATCH * NHEADS * DSTATE * HEAD_DIM * 4);
  _Float16* KVt  = (_Float16*)carve((size_t)BATCH * NHEADS * DSTATE * HEAD_DIM * 2);
  _Float16* ygh  = xBCr;  // pre-conv buffer is dead after k_conv; reuse for gated acts

  k_transpose_x<<<dim3(LSEQ / 32, DIMC / 32, BATCH), 256, 0, stream>>>(x, Xh);

  int cvt_n = NPROJ_PAD * DIMC + DIMC * DIMC;
  k_cvt_weights<<<(cvt_n + 255) / 256, 256, 0, stream>>>(W_in, W_out, Winh, Woh);

  k_inproj<<<dim3(LSEQ / 16, BATCH), 256, 0, stream>>>(Xh, Winh, A_log, dt_bias,
                                                       zh, xBCr, dtA);

  k_conv<<<dim3(LSEQ, BATCH), CONV_CH, 0, stream>>>(xBCr, conv_w, conv_b, xBCc);

  int kv_n = BATCH * NHEADS * DSTATE * HEAD_DIM;
  k_zero<<<(kv_n + 255) / 256, 256, 0, stream>>>(KVf, kv_n);

  k_kv<<<dim3(8, NHEADS, BATCH), 256, 0, stream>>>(xBCc, dtA, KVf);

  k_kvcvt_t<<<(kv_n + 255) / 256, 256, 0, stream>>>(KVf, KVt, kv_n);

  k_y<<<dim3(LSEQ / 16, BATCH), 256, 0, stream>>>(xBCc, KVt, Dv, ln_w, ln_b, zh, ygh);

  k_out<<<dim3(LSEQ / 16, BATCH), 256, 0, stream>>>(ygh, Woh, out);
}
